// RSNet_16535624090413
// MI455X (gfx1250) — compile-verified, matrix operand order checked by reading, not run on hardware
//
#include <hip/hip_runtime.h>
#include <cstdint>
#include <cstddef>

// ---------------------------------------------------------------------------
// Types / constants
// ---------------------------------------------------------------------------
typedef __attribute__((ext_vector_type(16))) _Float16 v16h;
typedef __attribute__((ext_vector_type(8)))  _Float16 v8h;
typedef __attribute__((ext_vector_type(8)))  float    v8f;

constexpr int kB  = 16;          // batch
constexpr int kN  = 8192;        // points per batch
constexpr int kP  = kB * kN;     // 131072 total points
constexpr int kS  = 50;          // NUM_SLICES
constexpr int kSB = kS * kB;     // 800 (slice-major GRU "rows")

union Frag16 { v16h v; v8h h[2]; };

__device__ __forceinline__ v8f zero_v8f() {
  union { v8f v; float e[8]; } u;
#pragma unroll
  for (int i = 0; i < 8; ++i) u.e[i] = 0.0f;
  return u.v;
}

// Load a 16x32 f16 WMMA fragment (A layout; B uses the mirrored column layout,
// which with row-major [N][K] weights is the identical addressing pattern).
// ISA 16-bit A layout: lanes 0-15 -> rows 0-15, K 0..7 then 16..23;
// lanes 16-31 -> rows 0-15, K 8..15 then 24..31.
__device__ __forceinline__ v16h load_frag16(const _Float16* base, int ld) {
  const int lane = threadIdx.x & 31;
  const int row  = lane & 15;
  const int kb   = (lane & 16) ? 8 : 0;
  const _Float16* p = base + row * ld + kb;
  Frag16 f;
  f.h[0] = *(const v8h*)(p);
  f.h[1] = *(const v8h*)(p + 16);
  return f.v;
}

// Low 32 bits of a generic pointer to __shared__ = LDS byte offset
// (LDS aperture base has zero low 32 bits; flat->LDS mapping truncates).
__device__ __forceinline__ unsigned lds_addr(const void* p) {
  return (unsigned)(uintptr_t)p;
}

// ---------------------------------------------------------------------------
// f32 -> f16 conversion
// ---------------------------------------------------------------------------
__global__ void k_f2h(const float* __restrict__ s, _Float16* __restrict__ d, int n) {
  int i = blockIdx.x * 256 + threadIdx.x;
  if (i < n) d[i] = (_Float16)s[i];
}

// ---------------------------------------------------------------------------
// slice indices: per (axis,b) min over N then floor((v-min)/RES)
// ---------------------------------------------------------------------------
__global__ __launch_bounds__(256) void k_sliceidx(const float* __restrict__ x,
                                                  int* __restrict__ idx) {
  const int ax = blockIdx.x >> 4;        // blockIdx.x = ax*16 + b
  const int b  = blockIdx.x & 15;
  const float* v = x + ((size_t)b * 3 + ax) * kN;
  __shared__ float red[256];
  float mn = 1e30f;
  for (int n = threadIdx.x; n < kN; n += 256) mn = fminf(mn, v[n]);
  red[threadIdx.x] = mn;
  __syncthreads();
  for (int s = 128; s > 0; s >>= 1) {
    if (threadIdx.x < s) red[threadIdx.x] = fminf(red[threadIdx.x], red[threadIdx.x + s]);
    __syncthreads();
  }
  const float zmin = red[0];
  const float res  = 0.02f + 1e-05f;
  int* out = idx + ((size_t)ax * kB + b) * kN;
  for (int n = threadIdx.x; n < kN; n += 256)
    out[n] = (int)floorf((v[n] - zmin) / res);
}

// ---------------------------------------------------------------------------
// conv1: 3 -> 64, input (B,3,N) channel-major, output point-major f16 [P][64]
// ---------------------------------------------------------------------------
__global__ __launch_bounds__(256) void k_conv1(const float* __restrict__ x,
                                               const float* __restrict__ w,
                                               const float* __restrict__ b,
                                               _Float16* __restrict__ y) {
  const int p = blockIdx.x * 4 + (threadIdx.x >> 6);
  const int o = threadIdx.x & 63;
  const int bi = p >> 13;
  const int n  = p & (kN - 1);
  const float* xb = x + (size_t)bi * 3 * kN + n;
  float v = b[o] + xb[0] * w[o * 3 + 0] + xb[kN] * w[o * 3 + 1] + xb[2 * kN] * w[o * 3 + 2];
  y[(size_t)p * 64 + o] = (_Float16)v;
}

// ---------------------------------------------------------------------------
// Generic WMMA GEMM: Y[M][N] = X[M][K] * W[N][K]^T + bias[N]
// block tile 128x64, BK=32, 8 waves (wave w -> M rows [16w,16w+16)).
// Tiles are staged global -> LDS with GLOBAL_LOAD_ASYNC_TO_LDS_B128 (ASYNCcnt),
// then consumed by v_wmma_f32_16x16x32_f16.
// Out-of-range A rows are clamped to row M-1: their products only reach
// D rows >= M, which the epilogue masks out.
// ---------------------------------------------------------------------------
template<bool OUT_HALF>
__global__ __launch_bounds__(256) void k_gemm(const _Float16* __restrict__ X,
                                              const _Float16* __restrict__ W,
                                              const float* __restrict__ bias,
                                              void* __restrict__ Yv,
                                              int M, int N, int K, int ldY, int Nstore) {
  (void)N;
  __shared__ __align__(16) _Float16 sA[128 * 32];
  __shared__ __align__(16) _Float16 sB[64 * 32];
  const int tid  = threadIdx.x;
  const int wave = tid >> 5;
  const int lane = tid & 31;
  const int m0 = blockIdx.x * 128;
  const int n0 = blockIdx.y * 64;

  v8f acc[4];
#pragma unroll
  for (int j = 0; j < 4; ++j) acc[j] = zero_v8f();

  const int rowA = tid >> 1;            // 0..127
  const int colA = (tid & 1) * 16;      // 0 / 16
  const int rowB = tid >> 2;            // 0..63
  const int colB = (tid & 3) * 8;       // 0/8/16/24

  const int gm  = m0 + rowA;
  const int gmc = gm < M ? gm : (M - 1);            // clamp tail rows
  const unsigned la0 = lds_addr(sA + rowA * 32 + colA);
  const unsigned la1 = la0 + 16u;
  const unsigned lb0 = lds_addr(sB + rowB * 32 + colB);

  for (int k0 = 0; k0 < K; k0 += 32) {
    const _Float16* srcA = X + (size_t)gmc * K + k0 + colA;
    const _Float16* srcB = W + (size_t)(n0 + rowB) * K + k0 + colB;
    // async tile staging: 2x16B of A + 1x16B of B per lane
    asm volatile("global_load_async_to_lds_b128 %0, %2, off\n\t"
                 "global_load_async_to_lds_b128 %1, %3, off"
                 :
                 : "v"(la0), "v"(la1),
                   "v"((unsigned long long)(uintptr_t)srcA),
                   "v"((unsigned long long)(uintptr_t)(srcA + 8))
                 : "memory");
    asm volatile("global_load_async_to_lds_b128 %0, %1, off"
                 :
                 : "v"(lb0),
                   "v"((unsigned long long)(uintptr_t)srcB)
                 : "memory");
    if (k0 + 32 < K) __builtin_prefetch(srcA + 32, 0, 1);   // global_prefetch_b8
    asm volatile("s_wait_asynccnt 0x0" ::: "memory");
    __syncthreads();

    v16h af = load_frag16(sA + wave * 16 * 32, 32);
#pragma unroll
    for (int j = 0; j < 4; ++j) {
      v16h bf = load_frag16(sB + j * 16 * 32, 32);
      acc[j] = __builtin_amdgcn_wmma_f32_16x16x32_f16(false, af, false, bf,
                                                      (short)0, acc[j], false, false);
    }
    __syncthreads();
  }

  // epilogue: C/D layout -> element (m = i + 8*(lane>=16), n = lane&15)
#pragma unroll
  for (int j = 0; j < 4; ++j) {
    const int col = n0 + j * 16 + (lane & 15);
    if (col < Nstore) {
      const float bv = bias ? bias[col] : 0.0f;
#pragma unroll
      for (int i = 0; i < 8; ++i) {
        const int m = m0 + wave * 16 + (lane >> 4) * 8 + i;
        if (m < M) {
          const float val = acc[j][i] + bv;
          if (OUT_HALF) ((_Float16*)Yv)[(size_t)m * ldY + col] = (_Float16)val;
          else          ((float*)Yv)[(size_t)m * ldY + col] = val;
        }
      }
    }
  }
}

// ---------------------------------------------------------------------------
// BatchNorm: per-channel sum/sumsq (atomics), then normalize + ReLU in place
// ---------------------------------------------------------------------------
__global__ __launch_bounds__(256) void k_bnstats(const _Float16* __restrict__ y,
                                                 float* __restrict__ sums,
                                                 float* __restrict__ sumsq, int C) {
  const int gid = blockIdx.x * 256 + threadIdx.x;
  const int c = gid % C;
  const int r0 = gid / C;
  const int rstep = (gridDim.x * 256) / C;
  float s = 0.0f, q = 0.0f;
  for (int r = r0; r < kP; r += rstep) {
    float v = (float)y[(size_t)r * C + c];
    s += v; q += v * v;
  }
  atomicAdd(&sums[c], s);
  atomicAdd(&sumsq[c], q);
}

__global__ __launch_bounds__(256) void k_bnapply(_Float16* __restrict__ y,
                                                 const float* __restrict__ sums,
                                                 const float* __restrict__ sumsq,
                                                 const float* __restrict__ g,
                                                 const float* __restrict__ b,
                                                 int C, int total) {
  const int stride = gridDim.x * 256;
  const float invP = 1.0f / (float)kP;
  for (int i = blockIdx.x * 256 + threadIdx.x; i < total; i += stride) {
    const int c = i % C;
    const float m  = sums[c] * invP;
    const float vr = sumsq[c] * invP - m * m;
    const float sc = g[c] * rsqrtf(vr + 1e-5f);
    const float v  = ((float)y[i] - m) * sc + b[c];
    y[i] = (_Float16)fmaxf(v, 0.0f);
  }
}

// ---------------------------------------------------------------------------
// segment-max pool: feat [P][64] (>=0 post-ReLU) -> pool[ax][(s*16+b)*64+c]
// uint-image atomicMax is order preserving for non-negative floats.
// ---------------------------------------------------------------------------
__global__ __launch_bounds__(64) void k_pool(const _Float16* __restrict__ feat,
                                             const int* __restrict__ idx,
                                             unsigned int* __restrict__ pl) {
  const int p = blockIdx.x;
  const int ax = blockIdx.y;
  const int c = threadIdx.x;
  const int b = p >> 13;
  const int s = idx[(size_t)ax * kP + p];
  const float v = (float)feat[(size_t)p * 64 + c];
  atomicMax(pl + ((size_t)ax * kSB + s * kB + b) * 64 + c, __float_as_uint(v));
}

// ---------------------------------------------------------------------------
// unpool + sum over the 3 axes -> rnn f16 [P][512]
// ---------------------------------------------------------------------------
__global__ __launch_bounds__(256) void k_unpool(const _Float16* __restrict__ sx,
                                                const _Float16* __restrict__ sy,
                                                const _Float16* __restrict__ sz,
                                                const int* __restrict__ idx,
                                                _Float16* __restrict__ rnn) {
  const int p = blockIdx.x;
  const int b = p >> 13;
  const int m0 = idx[p] * kB + b;
  const int m1 = idx[kP + p] * kB + b;
  const int m2 = idx[2 * kP + p] * kB + b;
  for (int c = threadIdx.x; c < 512; c += 256) {
    float v = (float)sx[(size_t)m0 * 512 + c] +
              (float)sy[(size_t)m1 * 512 + c] +
              (float)sz[(size_t)m2 * 512 + c];
    rnn[(size_t)p * 512 + c] = (_Float16)v;
  }
}

// ---------------------------------------------------------------------------
// GRU scan: one workgroup per direction; 50 sequential steps.
// Per step: gh[16][3H] = h[16][H] @ wh[3H][H]^T (WMMA), then gates in f32.
// xp[S*16][6H]: forward gates in cols [0,3H), backward in [3H,6H).
// out[S*16][2H]: forward -> cols [0,H), backward -> [H,2H).
// ---------------------------------------------------------------------------
template<int H>
__global__ __launch_bounds__(H == 64 ? 128 : 256)
void k_gru_scan(const float* __restrict__ xp,
                const _Float16* __restrict__ wh_f,
                const _Float16* __restrict__ wh_b,
                const float* __restrict__ bh_f,
                const float* __restrict__ bh_b,
                _Float16* __restrict__ out) {
  constexpr int NT = (3 * H) / 16;               // N tiles of the 16x3H result
  constexpr int WAVES = (H == 64) ? 4 : 8;
  constexpr int THREADS = WAVES * 32;
  constexpr int NJ = NT / WAVES;                 // tiles per wave: 3,3,6
  constexpr int KT = H / 32;
  __shared__ __align__(16) _Float16 h_lds[16 * H];
  __shared__ float gh[16 * 3 * H];

  const int tid  = threadIdx.x;
  const int wave = tid >> 5;
  const int lane = tid & 31;
  const int dir  = blockIdx.x;                   // 0 fwd, 1 bwd
  const _Float16* wh = dir ? wh_b : wh_f;
  const float*    bh = dir ? bh_b : bh_f;
  const int goff = dir ? 3 * H : 0;

  for (int i = tid; i < 16 * H; i += THREADS) h_lds[i] = (_Float16)0.0f;
  __syncthreads();

  for (int step = 0; step < kS; ++step) {
    const int t = dir ? (kS - 1 - step) : step;

    v8f acc[NJ];
#pragma unroll
    for (int jj = 0; jj < NJ; ++jj) acc[jj] = zero_v8f();

#pragma unroll
    for (int kt = 0; kt < KT; ++kt) {
      v16h af = load_frag16(h_lds + kt * 32, H);
#pragma unroll
      for (int jj = 0; jj < NJ; ++jj) {
        const int j = wave * NJ + jj;
        v16h bf = load_frag16(wh + (size_t)(j * 16) * H + kt * 32, H);
        acc[jj] = __builtin_amdgcn_wmma_f32_16x16x32_f16(false, af, false, bf,
                                                         (short)0, acc[jj], false, false);
      }
    }
#pragma unroll
    for (int jj = 0; jj < NJ; ++jj) {
      const int col = (wave * NJ + jj) * 16 + (lane & 15);
#pragma unroll
      for (int i = 0; i < 8; ++i) {
        const int m = (lane >> 4) * 8 + i;
        gh[m * 3 * H + col] = acc[jj][i];
      }
    }
    __syncthreads();

    // gates (each (b, j) element owned by exactly one thread)
    for (int e = tid; e < 16 * H; e += THREADS) {
      const int bi = e / H;
      const int jx = e % H;
      const float* xrow = xp + (size_t)(t * 16 + bi) * (6 * H) + goff;
      const float xr = xrow[jx];
      const float xz = xrow[H + jx];
      const float xn = xrow[2 * H + jx];
      const float hr = gh[bi * 3 * H + jx]         + bh[jx];
      const float hz = gh[bi * 3 * H + H + jx]     + bh[H + jx];
      const float hn = gh[bi * 3 * H + 2 * H + jx] + bh[2 * H + jx];
      const float r  = 1.0f / (1.0f + __expf(-(xr + hr)));
      const float z  = 1.0f / (1.0f + __expf(-(xz + hz)));
      const float nn = tanhf(xn + r * hn);
      const float hp = (float)h_lds[bi * H + jx];
      const float h2 = (1.0f - z) * nn + z * hp;
      h_lds[bi * H + jx] = (_Float16)h2;
      out[(size_t)(t * 16 + bi) * (2 * H) + dir * H + jx] = (_Float16)h2;
    }
    __syncthreads();
  }
}

// ---------------------------------------------------------------------------
// Host orchestration
// ---------------------------------------------------------------------------
extern "C" void kernel_launch(void* const* d_in, const int* in_sizes, int n_in,
                              void* d_out, int out_size, void* d_ws, size_t ws_size,
                              hipStream_t stream) {
  (void)in_sizes; (void)n_in; (void)out_size; (void)ws_size;
  static const int HL[6] = {256, 128, 64, 64, 128, 256};
  static const int FL[6] = {64, 512, 256, 128, 128, 256};

  // ---- input map (insertion order of setup_inputs dict) ----
  // 0:x 1:conv1_w 2:conv1_b 3:conv2_w 4:conv2_b 5:conv3_w 6:conv3_b
  // 7:bn1_g 8:bn1_b 9:bn2_g 10:bn2_b 11:bn3_g 12:bn3_b
  // 13:conv6_w 14:conv6_b 15:bn6_g 16:bn6_b 17:conv7_w 18:conv7_b 19:bn7_g 20:bn7_b
  // 21:conv8_w 22:conv8_b
  // 23 + ax*48 + l*8 + k ; k: 0 wi_f,1 wh_f,2 bi_f,3 bh_f,4 wi_b,5 wh_b,6 bi_b,7 bh_b
  auto IN = [&](int i) { return (const float*)d_in[i]; };
  auto G  = [&](int ax, int l, int k) { return 23 + ax * 48 + l * 8 + k; };
  const float* x_in = IN(0);

  // ---- workspace layout ----
  char* ws = (char*)d_ws;
  size_t off = 0;
  auto alloc = [&](size_t bytes) {
    size_t o = off;
    off += (bytes + 255) & ~(size_t)255;
    return o;
  };
  size_t o_idx  = alloc((size_t)3 * kP * sizeof(int));
  size_t o_actA = alloc((size_t)kP * 512 * sizeof(_Float16));
  size_t o_actB = alloc((size_t)kP * 512 * sizeof(_Float16));
  size_t o_sum  = alloc(512 * sizeof(float));
  size_t o_sq   = alloc(512 * sizeof(float));
  size_t o_pool = alloc((size_t)3 * kSB * 64 * sizeof(float));
  size_t o_seqA = alloc((size_t)3 * kSB * 512 * sizeof(_Float16));
  size_t o_seqB = alloc((size_t)3 * kSB * 512 * sizeof(_Float16));
  size_t o_xp   = alloc((size_t)3 * kSB * 1536 * sizeof(float));
  size_t o_w2   = alloc((size_t)64 * 64 * sizeof(_Float16));
  size_t o_w3   = alloc((size_t)64 * 64 * sizeof(_Float16));
  size_t o_w6   = alloc((size_t)512 * 512 * sizeof(_Float16));
  size_t o_w7   = alloc((size_t)256 * 512 * sizeof(_Float16));
  size_t o_w8   = alloc((size_t)64 * 256 * sizeof(_Float16));
  size_t o_b8   = alloc(64 * sizeof(float));
  size_t o_wi[3][6], o_whf[3][6], o_whb[3][6], o_bic[3][6];
  for (int ax = 0; ax < 3; ++ax)
    for (int l = 0; l < 6; ++l) {
      const int H = HL[l], F = FL[l];
      o_wi[ax][l]  = alloc((size_t)6 * H * F * sizeof(_Float16));
      o_whf[ax][l] = alloc((size_t)3 * H * H * sizeof(_Float16));
      o_whb[ax][l] = alloc((size_t)3 * H * H * sizeof(_Float16));
      o_bic[ax][l] = alloc((size_t)6 * H * sizeof(float));
    }

  int*       idx  = (int*)(ws + o_idx);
  _Float16*  actA = (_Float16*)(ws + o_actA);
  _Float16*  actB = (_Float16*)(ws + o_actB);
  float*     sums = (float*)(ws + o_sum);
  float*     sq   = (float*)(ws + o_sq);
  float*     pool = (float*)(ws + o_pool);
  _Float16*  seqA = (_Float16*)(ws + o_seqA);
  _Float16*  seqB = (_Float16*)(ws + o_seqB);

  auto cvt = [&](const float* src, _Float16* dst, int n) {
    k_f2h<<<dim3((n + 255) / 256), dim3(256), 0, stream>>>(src, dst, n);
  };
  auto gemm = [&](bool outHalf, const _Float16* X, const _Float16* W, const float* bias,
                  void* Y, int M, int N, int K, int ldY, int Nstore) {
    dim3 g((M + 127) / 128, N / 64);
    if (outHalf) k_gemm<true ><<<g, 256, 0, stream>>>(X, W, bias, Y, M, N, K, ldY, Nstore);
    else         k_gemm<false><<<g, 256, 0, stream>>>(X, W, bias, Y, M, N, K, ldY, Nstore);
  };
  auto bnorm = [&](_Float16* y, int C, const float* g_, const float* b_) {
    hipMemsetAsync(sums, 0, 512 * sizeof(float), stream);
    hipMemsetAsync(sq,   0, 512 * sizeof(float), stream);
    k_bnstats<<<1024, 256, 0, stream>>>(y, sums, sq, C);
    k_bnapply<<<2048, 256, 0, stream>>>(y, sums, sq, g_, b_, C, kP * C);
  };

  // ---- weight preparation (f32 -> f16, concatenations, padding) ----
  cvt(IN(3),  (_Float16*)(ws + o_w2), 64 * 64);
  cvt(IN(5),  (_Float16*)(ws + o_w3), 64 * 64);
  cvt(IN(13), (_Float16*)(ws + o_w6), 512 * 512);
  cvt(IN(17), (_Float16*)(ws + o_w7), 256 * 512);
  hipMemsetAsync(ws + o_w8, 0, (size_t)64 * 256 * sizeof(_Float16), stream);
  cvt(IN(21), (_Float16*)(ws + o_w8), 13 * 256);
  hipMemsetAsync(ws + o_b8, 0, 64 * sizeof(float), stream);
  hipMemcpyAsync(ws + o_b8, d_in[22], 13 * sizeof(float), hipMemcpyDeviceToDevice, stream);
  for (int ax = 0; ax < 3; ++ax)
    for (int l = 0; l < 6; ++l) {
      const int H = HL[l], F = FL[l], G3 = 3 * H;
      _Float16* wi = (_Float16*)(ws + o_wi[ax][l]);
      cvt(IN(G(ax, l, 0)), wi, G3 * F);                      // wi_f -> cols [0,3H)
      cvt(IN(G(ax, l, 4)), wi + (size_t)G3 * F, G3 * F);     // wi_b -> cols [3H,6H)
      cvt(IN(G(ax, l, 1)), (_Float16*)(ws + o_whf[ax][l]), G3 * H);
      cvt(IN(G(ax, l, 5)), (_Float16*)(ws + o_whb[ax][l]), G3 * H);
      hipMemcpyAsync(ws + o_bic[ax][l], d_in[G(ax, l, 2)],
                     (size_t)G3 * sizeof(float), hipMemcpyDeviceToDevice, stream);
      hipMemcpyAsync(ws + o_bic[ax][l] + (size_t)G3 * sizeof(float), d_in[G(ax, l, 6)],
                     (size_t)G3 * sizeof(float), hipMemcpyDeviceToDevice, stream);
    }

  // ---- pipeline ----
  k_sliceidx<<<48, 256, 0, stream>>>(x_in, idx);
  k_conv1<<<kP / 4, 256, 0, stream>>>(x_in, IN(1), IN(2), actA);
  bnorm(actA, 64, IN(7), IN(8));
  gemm(true, actA, (_Float16*)(ws + o_w2), IN(4), actB, kP, 64, 64, 64, 64);
  bnorm(actB, 64, IN(9), IN(10));
  gemm(true, actB, (_Float16*)(ws + o_w3), IN(6), actA, kP, 64, 64, 64, 64);
  bnorm(actA, 64, IN(11), IN(12));

  hipMemsetAsync(pool, 0, (size_t)3 * kSB * 64 * sizeof(float), stream);
  k_pool<<<dim3(kP, 3), 64, 0, stream>>>(actA, idx, (unsigned int*)pool);
  // pooled f32 [800][64] -> f16 seq (contiguous [800][64] at the head of each axis slot)
  for (int ax = 0; ax < 3; ++ax)
    cvt(pool + (size_t)ax * kSB * 64, seqA + (size_t)ax * kSB * 512, kSB * 64);

  for (int l = 0; l < 6; ++l) {
    const int H = HL[l], F = FL[l];
    for (int ax = 0; ax < 3; ++ax) {
      _Float16* sin  = ((l & 1) == 0 ? seqA : seqB) + (size_t)ax * kSB * 512;
      _Float16* sout = ((l & 1) == 0 ? seqB : seqA) + (size_t)ax * kSB * 512;
      float* xpax = (float*)(ws + o_xp) + (size_t)ax * kSB * 1536;
      gemm(false, sin, (_Float16*)(ws + o_wi[ax][l]),
           (const float*)(ws + o_bic[ax][l]), xpax, kSB, 6 * H, F, 6 * H, 6 * H);
      const _Float16* whf = (const _Float16*)(ws + o_whf[ax][l]);
      const _Float16* whb = (const _Float16*)(ws + o_whb[ax][l]);
      const float* bhf = IN(G(ax, l, 3));
      const float* bhb = IN(G(ax, l, 7));
      if (H == 256)
        k_gru_scan<256><<<2, 256, 0, stream>>>(xpax, whf, whb, bhf, bhb, sout);
      else if (H == 128)
        k_gru_scan<128><<<2, 256, 0, stream>>>(xpax, whf, whb, bhf, bhb, sout);
      else
        k_gru_scan<64><<<2, 128, 0, stream>>>(xpax, whf, whb, bhf, bhb, sout);
    }
  }

  // layer 5 writes seqA; unpool + sum into actA (h3 no longer needed)
  k_unpool<<<kP, 256, 0, stream>>>(seqA + (size_t)0 * kSB * 512,
                                   seqA + (size_t)1 * kSB * 512,
                                   seqA + (size_t)2 * kSB * 512, idx, actA);

  gemm(true, actA, (_Float16*)(ws + o_w6), IN(14), actB, kP, 512, 512, 512, 512);
  bnorm(actB, 512, IN(15), IN(16));
  gemm(true, actB, (_Float16*)(ws + o_w7), IN(18), actA, kP, 256, 512, 256, 256);
  bnorm(actA, 256, IN(19), IN(20));
  // conv8 (weights/bias zero-padded to N=64); writes f32 straight into d_out (B,N,13)
  gemm(false, actA, (_Float16*)(ws + o_w8), (const float*)(ws + o_b8),
       d_out, kP, 64, 256, 13, 13);
}
